// ContinuousGaussian_34024730919751
// MI455X (gfx1250) — matrix operand order repaired and checked by reading, not compiled, not run on hardware
//
#include <hip/hip_runtime.h>
#include <hip/hip_bf16.h>

typedef __bf16 bf16;
typedef __attribute__((ext_vector_type(16))) __bf16 v16bf;
typedef __attribute__((ext_vector_type(8)))  float  v8f;

// ---------------------------------------------------------------------------
// CDNA5 async global->LDS copy helpers (ASYNCcnt-tracked)
// ---------------------------------------------------------------------------
__device__ __forceinline__ void async_b128(void* lds, const void* g) {
    asm volatile("global_load_async_to_lds_b128 %0, %1, off"
                 :: "v"((unsigned)(uintptr_t)lds), "v"(g) : "memory");
}
__device__ __forceinline__ void wait_async() {
#if __has_builtin(__builtin_amdgcn_s_wait_asynccnt)
    __builtin_amdgcn_s_wait_asynccnt(0);
#else
    asm volatile("s_wait_asynccnt 0" ::: "memory");
#endif
}

// ---------------------------------------------------------------------------
// small prep kernels
// ---------------------------------------------------------------------------
__global__ void k_f2b(const float* __restrict__ s, bf16* __restrict__ d, int n) {
    int i = blockIdx.x * 256 + threadIdx.x;
    if (i < n) d[i] = (bf16)s[i];
}

// W[K][N] f32 -> Wt[N][K] bf16  (one-time, so all GEMM B-tiles are [N][K])
__global__ void k_f2bT(const float* __restrict__ s, bf16* __restrict__ d, int K, int N) {
    int i = blockIdx.x * 256 + threadIdx.x;
    if (i >= K * N) return;
    int k = i / N, n = i - k * N;
    d[(size_t)n * K + k] = (bf16)s[i];
}

// feat (2,256,64,64) -> x_bf16[n][c], xo_bf16[n][c] (leaky), pc_bf16 NCHW (leaky)
__global__ void k_prep_feat(const float* __restrict__ feat, bf16* __restrict__ xb,
                            bf16* __restrict__ xob, bf16* __restrict__ pcb) {
    int i = blockIdx.x * 256 + threadIdx.x;   // over 2*256*4096
    if (i >= 2 * 256 * 4096) return;
    int hw = i & 4095;
    int c  = (i >> 12) & 255;
    int b  = i >> 20;
    float v  = feat[i];
    float pv = v > 0.f ? v : 0.01f * v;
    pcb[i] = (bf16)pv;
    int n = (b << 12) | hw;
    xb[n * 256 + c]  = (bf16)v;
    xob[n * 256 + c] = (bf16)pv;
}

// gaussian dictionary: 9*9*9 + zero row = 730 x 3
__global__ void k_gau(float* __restrict__ gF, bf16* __restrict__ gB) {
    const float c1[9] = {0.0f, 0.41f, 0.62f, 0.98f, 1.13f, 1.29f, 1.64f, 1.85f, 2.36f};
    const float c2[9] = {-0.86f, -0.36f, -0.16f, 0.19f, 0.34f, 0.49f, 0.84f, 1.04f, 1.54f};
    const float c3[9] = {0.0f, 0.33f, 0.53f, 0.88f, 1.03f, 1.18f, 1.53f, 1.73f, 2.23f};
    int id = blockIdx.x * 256 + threadIdx.x;
    if (id >= 730) return;
    float a = 0.f, b = 0.f, c = 0.f;
    if (id < 729) {
        int i = id / 81, j = (id / 9) % 9, k = id % 9;
        a = c1[i]; b = c2[j]; c = c3[k];
    }
    gF[id * 3 + 0] = a; gF[id * 3 + 1] = b; gF[id * 3 + 2] = c;
    gB[id * 3 + 0] = (bf16)a; gB[id * 3 + 1] = (bf16)b; gB[id * 3 + 2] = (bf16)c;
}

// ---------------------------------------------------------------------------
// bf16 WMMA GEMM: out = act(A[M,K] @ Bt^T + bias), Bt given as [N][K] row-major.
// Block: 256 thr = 8 waves; tile 64x128; wave tile 32x32 (4 accumulators).
// Double-buffered LDS; fast-path fills via global_load_async_to_lds_b128.
// ---------------------------------------------------------------------------
__launch_bounds__(256)
__global__ void k_gemm(const bf16* __restrict__ A, const bf16* __restrict__ Bt,
                       const float* __restrict__ bias,
                       float* __restrict__ outF, bf16* __restrict__ outH,
                       int M, int N, int K, int act) {
    __shared__ __align__(16) bf16 As[2][64 * 32];    // [m][k]
    __shared__ __align__(16) bf16 Bs[2][128 * 32];   // [n][k]
    const int tid  = threadIdx.x;
    const int lane = tid & 31;
    const int wv   = tid >> 5;      // 0..7
    const int wm   = wv & 1;        // 2 row slabs of 32
    const int wn   = wv >> 1;       // 4 col slabs of 32
    const int mr   = lane & 15;
    const int hh   = lane >> 4;
    const int m0 = blockIdx.y * 64;
    const int n0 = blockIdx.x * 128;
    v8f acc[2][2] = {{{}, {}}, {{}, {}}};

    auto fill = [&](int buf, int k0) {
        { // A tile 64x32
            int r = tid >> 2, c0 = (tid & 3) * 8;
            if ((m0 + 64 <= M) && (k0 + 32 <= K)) {
                async_b128(&As[buf][r * 32 + c0], &A[(size_t)(m0 + r) * K + k0 + c0]);
            } else {
                int row = m0 + r;
                int rowc = row < M ? row : (M - 1);
                #pragma unroll
                for (int i = 0; i < 8; ++i) {
                    int k = k0 + c0 + i;
                    int kc = k < K ? k : (K - 1);
                    bf16 vv = A[(size_t)rowc * K + kc];
                    As[buf][r * 32 + c0 + i] = (row < M && k < K) ? vv : (bf16)0.f;
                }
            }
        }
        { // B tile 128x32 (straight copy from [N][K])
            int r = tid >> 1, c0 = (tid & 1) * 16;
            if ((n0 + 128 <= N) && (k0 + 32 <= K)) {
                const bf16* src = &Bt[(size_t)(n0 + r) * K + k0 + c0];
                async_b128(&Bs[buf][r * 32 + c0],     src);
                async_b128(&Bs[buf][r * 32 + c0 + 8], src + 8);
            } else {
                int col = n0 + r;
                int colc = col < N ? col : (N - 1);
                #pragma unroll
                for (int i = 0; i < 16; ++i) {
                    int k = k0 + c0 + i;
                    int kc = k < K ? k : (K - 1);
                    bf16 vv = Bt[(size_t)colc * K + kc];
                    Bs[buf][r * 32 + c0 + i] = (col < N && k < K) ? vv : (bf16)0.f;
                }
            }
        }
    };

    fill(0, 0);
    int pb = 0;
    for (int k0 = 0; k0 < K; k0 += 32) {
        wait_async();
        __syncthreads();
        if (k0 + 32 < K) fill(pb ^ 1, k0 + 32);

        // fragments per ISA 7.12.2 16-bit layouts
        v16bf a[2], b[2];
        #pragma unroll
        for (int mi = 0; mi < 2; ++mi) {
            #pragma unroll
            for (int v = 0; v < 8; ++v) {
                int kb = ((v < 4) ? 0 : 16) + 8 * hh + 2 * (v & 3);
                int base = (wm * 32 + mi * 16 + mr) * 32 + kb;
                a[mi][2 * v]     = As[pb][base];
                a[mi][2 * v + 1] = As[pb][base + 1];
            }
        }
        #pragma unroll
        for (int ni = 0; ni < 2; ++ni) {
            #pragma unroll
            for (int v = 0; v < 8; ++v) {
                int kb = ((v < 4) ? 0 : 16) + 8 * hh + 2 * (v & 3);
                int base = (wn * 32 + ni * 16 + mr) * 32 + kb;
                b[ni][2 * v]     = Bs[pb][base];
                b[ni][2 * v + 1] = Bs[pb][base + 1];
            }
        }
        #pragma unroll
        for (int mi = 0; mi < 2; ++mi)
            #pragma unroll
            for (int ni = 0; ni < 2; ++ni)
                acc[mi][ni] = __builtin_amdgcn_wmma_f32_16x16x32_bf16(
                    false, a[mi], false, b[ni], (short)0, acc[mi][ni], false, false);
        pb ^= 1;
    }

    // epilogue: D layout — VGPR r holds (M = r + 8*hh, N = lane&15)
    #pragma unroll
    for (int mi = 0; mi < 2; ++mi) {
        #pragma unroll
        for (int r = 0; r < 8; ++r) {
            int row = m0 + wm * 32 + mi * 16 + r + 8 * hh;
            #pragma unroll
            for (int ni = 0; ni < 2; ++ni) {
                int col = n0 + wn * 32 + ni * 16 + mr;
                if (row < M && col < N) {
                    float v = acc[mi][ni][r];
                    if (bias) v += bias[col];
                    if (act == 1) v = v > 0.f ? v : 0.f;
                    else if (act == 2) v = tanhf(v);
                    if (outF) outF[(size_t)row * N + col] = v;
                    if (outH) outH[(size_t)row * N + col] = (bf16)v;
                }
            }
        }
    }
}

// ---------------------------------------------------------------------------
// conv1 3x3 SAME as implicit GEMM: M=8192 pixels, N=512, K=2304 (=256*9)
// wTb = conv1_w flat as [oc][k] bf16 (already B^T layout). Double-buffered;
// B fill async, A tile = branchless on-the-fly im2col.
// ---------------------------------------------------------------------------
__launch_bounds__(256)
__global__ void k_conv_gemm(const bf16* __restrict__ pcb, const bf16* __restrict__ wTb,
                            const float* __restrict__ bias, bf16* __restrict__ outH) {
    __shared__ __align__(16) bf16 As[2][64 * 32];
    __shared__ __align__(16) bf16 Bs[2][128 * 32];
    const int tid  = threadIdx.x;
    const int lane = tid & 31;
    const int wv   = tid >> 5;
    const int wm   = wv & 1;
    const int wn   = wv >> 1;
    const int mr   = lane & 15;
    const int hh   = lane >> 4;
    const int m0 = blockIdx.y * 64;
    const int n0 = blockIdx.x * 128;
    const int K = 2304, Nc = 512;
    v8f acc[2][2] = {{{}, {}}, {{}, {}}};

    auto fill = [&](int buf, int k0) {
        { // im2col A tile (branchless: clamp + select)
            int r = tid >> 2, c0 = (tid & 3) * 8;
            int row = m0 + r;
            int bb = row >> 12, hw = row & 4095, y = hw >> 6, x = hw & 63;
            #pragma unroll
            for (int i = 0; i < 8; ++i) {
                int k  = k0 + c0 + i;
                int ic = k / 9, r9 = k - ic * 9;
                int q3 = r9 / 3;
                int ky = q3 - 1;
                int kx = r9 - q3 * 3 - 1;
                int yy = y + ky, xx = x + kx;
                bool in = (yy >= 0) & (yy < 64) & (xx >= 0) & (xx < 64);
                int yc = yy < 0 ? 0 : (yy > 63 ? 63 : yy);
                int xc = xx < 0 ? 0 : (xx > 63 ? 63 : xx);
                bf16 vv = pcb[(((size_t)(bb * 256 + ic)) << 12) + (yc << 6) + xc];
                As[buf][r * 32 + c0 + i] = in ? vv : (bf16)0.f;
            }
        }
        { // B tile: straight async copy from [oc][k]
            int r = tid >> 1, c0 = (tid & 1) * 16;
            const bf16* src = &wTb[(size_t)(n0 + r) * K + k0 + c0];
            async_b128(&Bs[buf][r * 32 + c0],     src);
            async_b128(&Bs[buf][r * 32 + c0 + 8], src + 8);
        }
    };

    fill(0, 0);
    int pb = 0;
    for (int k0 = 0; k0 < K; k0 += 32) {
        wait_async();
        __syncthreads();
        if (k0 + 32 < K) fill(pb ^ 1, k0 + 32);

        v16bf a[2], b[2];
        #pragma unroll
        for (int mi = 0; mi < 2; ++mi) {
            #pragma unroll
            for (int v = 0; v < 8; ++v) {
                int kb = ((v < 4) ? 0 : 16) + 8 * hh + 2 * (v & 3);
                int base = (wm * 32 + mi * 16 + mr) * 32 + kb;
                a[mi][2 * v]     = As[pb][base];
                a[mi][2 * v + 1] = As[pb][base + 1];
            }
        }
        #pragma unroll
        for (int ni = 0; ni < 2; ++ni) {
            #pragma unroll
            for (int v = 0; v < 8; ++v) {
                int kb = ((v < 4) ? 0 : 16) + 8 * hh + 2 * (v & 3);
                int base = (wn * 32 + ni * 16 + mr) * 32 + kb;
                b[ni][2 * v]     = Bs[pb][base];
                b[ni][2 * v + 1] = Bs[pb][base + 1];
            }
        }
        #pragma unroll
        for (int mi = 0; mi < 2; ++mi)
            #pragma unroll
            for (int ni = 0; ni < 2; ++ni)
                acc[mi][ni] = __builtin_amdgcn_wmma_f32_16x16x32_bf16(
                    false, a[mi], false, b[ni], (short)0, acc[mi][ni], false, false);
        pb ^= 1;
    }
    #pragma unroll
    for (int mi = 0; mi < 2; ++mi) {
        #pragma unroll
        for (int r = 0; r < 8; ++r) {
            int row = m0 + wm * 32 + mi * 16 + r + 8 * hh;
            #pragma unroll
            for (int ni = 0; ni < 2; ++ni) {
                int col = n0 + wn * 32 + ni * 16 + mr;
                float v = acc[mi][ni][r] + bias[col];
                outH[(size_t)row * Nc + col] = (bf16)v;
            }
        }
    }
}

// ---------------------------------------------------------------------------
// softmax over 730 gaussians per pixel column + para = wts^T @ gau
// ---------------------------------------------------------------------------
__launch_bounds__(256)
__global__ void k_softmax_para(const float* __restrict__ logits,
                               const float* __restrict__ gauF,
                               float* __restrict__ para) {
    __shared__ float g[730 * 3];
    for (int i = threadIdx.x; i < 730 * 3; i += 256) g[i] = gauF[i];
    __syncthreads();
    int n = blockIdx.x * 256 + threadIdx.x;
    float mx = -3.4e38f;
    for (int gi = 0; gi < 730; ++gi)
        mx = fmaxf(mx, logits[(size_t)gi * 8192 + n]);
    float s = 0.f, p0 = 0.f, p1 = 0.f, p2 = 0.f;
    for (int gi = 0; gi < 730; ++gi) {
        float e = __expf(logits[(size_t)gi * 8192 + n] - mx);
        s  += e;
        p0 += e * g[gi * 3 + 0];
        p1 += e * g[gi * 3 + 1];
        p2 += e * g[gi * 3 + 2];
    }
    float inv = 1.f / s;
    para[n * 3 + 0] = p0 * inv;
    para[n * 3 + 1] = p1 * inv;
    para[n * 3 + 2] = p2 * inv;
}

// ---------------------------------------------------------------------------
// splat: per batch, out[c][pix] = sum_g alpha(g,pix) * color[g][c]
// ---------------------------------------------------------------------------
__launch_bounds__(256)
__global__ void k_splat(const float* __restrict__ color, const float* __restrict__ para,
                        const float* __restrict__ off, float* __restrict__ out) {
    __shared__ float sc0[256], sc1[256], sc2[256];
    __shared__ float sa[256], sb[256], sq[256], sgx[256], sgy[256];
    int pix = blockIdx.x * 256 + threadIdx.x;   // 0..8191
    int b = pix >> 12;
    int p = pix & 4095;
    float px = (float)(p & 63);
    float py = (float)(p >> 6);
    float a0 = 0.f, a1 = 0.f, a2 = 0.f;
    for (int g0 = 0; g0 < 4096; g0 += 256) {
        int jj = g0 + threadIdx.x;
        int gi = (b << 12) + jj;
        {
            float jx = (float)(jj & 63);
            float jy = (float)(jj >> 6);
            sgx[threadIdx.x] = jx + off[gi * 2 + 0] - 0.5f;
            sgy[threadIdx.x] = jy + off[gi * 2 + 1] - 0.5f;
            float l1 = para[gi * 3 + 0] * 0.25f;
            float l2 = para[gi * 3 + 1] * 0.25f;
            float l3 = para[gi * 3 + 2] * 0.25f;
            float det = (l1 * l3) * (l1 * l3);
            float inv = 1.f / det;
            sa[threadIdx.x] = (l2 * l2 + l3 * l3) * inv;
            sb[threadIdx.x] = -(l1 * l2) * inv;
            sq[threadIdx.x] = (l1 * l1) * inv;
            sc0[threadIdx.x] = color[gi * 3 + 0];
            sc1[threadIdx.x] = color[gi * 3 + 1];
            sc2[threadIdx.x] = color[gi * 3 + 2];
        }
        __syncthreads();
        for (int t = 0; t < 256; ++t) {
            float dx = px - sgx[t];
            float dy = py - sgy[t];
            float sg = 0.5f * sa[t] * dx * dx + 0.5f * sq[t] * dy * dy + sb[t] * dx * dy;
            float al = (sg >= 0.f) ? fminf(0.99f, __expf(-sg)) : 0.f;
            a0 += al * sc0[t];
            a1 += al * sc1[t];
            a2 += al * sc2[t];
        }
        __syncthreads();
    }
    out[((b * 3 + 0) << 12) + p] = a0;
    out[((b * 3 + 1) << 12) + p] = a1;
    out[((b * 3 + 2) << 12) + p] = a2;
}

// ---------------------------------------------------------------------------
// host orchestration
// ---------------------------------------------------------------------------
extern "C" void kernel_launch(void* const* d_in, const int* in_sizes, int n_in,
                              void* d_out, int out_size, void* d_ws, size_t ws_size,
                              hipStream_t stream) {
    (void)in_sizes; (void)n_in; (void)out_size; (void)ws_size;
    const int N = 8192;   // bs*HW

    const float* feat  = (const float*)d_in[0];
    const float* convW = (const float*)d_in[1];
    const float* convB = (const float*)d_in[2];
    const float *Wc[6], *bc[6], *Wv[5], *bv[5], *Wo[6], *bo[6];
    for (int i = 0; i < 6; ++i) { Wc[i] = (const float*)d_in[3  + 2 * i]; bc[i] = (const float*)d_in[4  + 2 * i]; }
    for (int i = 0; i < 5; ++i) { Wv[i] = (const float*)d_in[15 + 2 * i]; bv[i] = (const float*)d_in[16 + 2 * i]; }
    for (int i = 0; i < 6; ++i) { Wo[i] = (const float*)d_in[25 + 2 * i]; bo[i] = (const float*)d_in[26 + 2 * i]; }

    static const int colK[6] = {256, 512, 1024, 256, 128, 64};
    static const int colN[6] = {512, 1024, 256, 128, 64, 3};
    static const int vecK[5] = {3, 256, 512, 512, 512};
    static const int vecN[5] = {256, 512, 512, 512, 512};
    static const int offN[6] = {512, 1024, 256, 128, 64, 2};

    size_t woff = 0;
    auto alloc = [&](size_t bytes) -> void* {
        woff = (woff + 255) & ~(size_t)255;
        void* p = (char*)d_ws + woff;
        woff += bytes;
        return p;
    };
    bf16*  xb     = (bf16*)alloc((size_t)N * 256 * 2);
    bf16*  xob    = (bf16*)alloc((size_t)N * 256 * 2);
    bf16*  pcb    = (bf16*)alloc((size_t)N * 256 * 2);
    float* gauF   = (float*)alloc(730 * 3 * 4);
    bf16*  gauB   = (bf16*)alloc(730 * 3 * 2);
    bf16 *wcb[6], *wvb[5], *wob[6];
    for (int i = 0; i < 6; ++i) wcb[i] = (bf16*)alloc((size_t)colK[i] * colN[i] * 2);
    for (int i = 0; i < 5; ++i) wvb[i] = (bf16*)alloc((size_t)vecK[i] * vecN[i] * 2);
    for (int i = 0; i < 6; ++i) wob[i] = (bf16*)alloc((size_t)colK[i] * offN[i] * 2);
    bf16*  convWb  = (bf16*)alloc((size_t)512 * 2304 * 2);
    bf16*  actA    = (bf16*)alloc((size_t)N * 1024 * 2);
    bf16*  actB    = (bf16*)alloc((size_t)N * 1024 * 2);
    bf16*  vecA    = (bf16*)alloc((size_t)730 * 512 * 2);
    bf16*  vecB    = (bf16*)alloc((size_t)730 * 512 * 2);
    bf16*  convOut = (bf16*)alloc((size_t)N * 512 * 2);
    float* logits  = (float*)alloc((size_t)730 * N * 4);
    float* colorF  = (float*)alloc((size_t)N * 3 * 4);
    float* paraF   = (float*)alloc((size_t)N * 3 * 4);
    float* offF    = (float*)alloc((size_t)N * 2 * 4);

    // weights -> [N][K] bf16 (one-time transpose-convert)
    auto cvtT = [&](const float* s, bf16* d, int K, int Nn) {
        k_f2bT<<<(K * Nn + 255) / 256, 256, 0, stream>>>(s, d, K, Nn);
    };
    for (int i = 0; i < 6; ++i) cvtT(Wc[i], wcb[i], colK[i], colN[i]);
    for (int i = 0; i < 5; ++i) cvtT(Wv[i], wvb[i], vecK[i], vecN[i]);
    for (int i = 0; i < 6; ++i) cvtT(Wo[i], wob[i], colK[i], offN[i]);
    // conv1_w flat (512,2304) is already [oc][k] = B^T: plain convert
    k_f2b<<<(512 * 2304 + 255) / 256, 256, 0, stream>>>(convW, convWb, 512 * 2304);
    k_prep_feat<<<(2 * 256 * 4096) / 256, 256, 0, stream>>>(feat, xb, xob, pcb);
    k_gau<<<3, 256, 0, stream>>>(gauF, gauB);

    auto gemm = [&](const bf16* A, const bf16* Bt, const float* bias,
                    float* oF, bf16* oH, int M, int Nn, int K, int act) {
        dim3 g((Nn + 127) / 128, (M + 63) / 64);
        k_gemm<<<g, 256, 0, stream>>>(A, Bt, bias, oF, oH, M, Nn, K, act);
    };

    // mlp_color: 256->512->1024->256->128->64->3 (ReLU x5, last linear -> fp32)
    {
        const bf16* cur = xb;
        bf16* bufs[2] = {actA, actB};
        for (int l = 0; l < 6; ++l) {
            if (l < 5) {
                bf16* nxt = bufs[l & 1];
                gemm(cur, wcb[l], bc[l], nullptr, nxt, N, colN[l], colK[l], 1);
                cur = nxt;
            } else {
                gemm(cur, wcb[l], bc[l], colorF, nullptr, N, colN[l], colK[l], 0);
            }
        }
    }
    // mlp_off: same dims, final tanh -> fp32
    {
        const bf16* cur = xob;
        bf16* bufs[2] = {actA, actB};
        for (int l = 0; l < 6; ++l) {
            if (l < 5) {
                bf16* nxt = bufs[l & 1];
                gemm(cur, wob[l], bo[l], nullptr, nxt, N, offN[l], colK[l], 1);
                cur = nxt;
            } else {
                gemm(cur, wob[l], bo[l], offF, nullptr, N, offN[l], colK[l], 2);
            }
        }
    }
    // mlp_vec on gau (730 rows), last layer linear -> bf16 "vector" (730x512)
    bf16* vector = nullptr;
    {
        const bf16* cur = gauB;
        bf16* bufs[2] = {vecA, vecB};
        for (int l = 0; l < 5; ++l) {
            bf16* nxt = bufs[l & 1];
            gemm(cur, wvb[l], bv[l], nullptr, nxt, 730, vecN[l], vecK[l], (l < 4) ? 1 : 0);
            cur = nxt;
            vector = nxt;
        }
    }
    // conv1 implicit GEMM -> convOut [pixel][oc] bf16
    k_conv_gemm<<<dim3(512 / 128, N / 64), 256, 0, stream>>>(pcb, convWb, convB, convOut);
    // logits = vector @ p : Bt = convOut [pixel][oc] = [N][K] -> fp32 (730 x 8192)
    gemm(vector, convOut, nullptr, logits, nullptr, 730, N, 512, 0);

    k_softmax_para<<<N / 256, 256, 0, stream>>>(logits, gauF, paraF);
    k_splat<<<N / 256, 256, 0, stream>>>(colorF, paraF, offF, (float*)d_out);
}
